// Gat_8967891714110
// MI455X (gfx1250) — matrix operand order repaired and checked by reading, not compiled, not run on hardware
//
#include <hip/hip_runtime.h>
#include <hip/hip_bf16.h>

typedef __attribute__((ext_vector_type(16))) _Float16 v16h;
typedef __attribute__((ext_vector_type(8)))  float    v8f;

#define NEG_SLOPE 0.2f
#define EPS_DEN   1e-16f
#define ENC_NEG_INF 0x007fffffu   // encF(-inf)

__device__ __forceinline__ unsigned encF(float f) {
  unsigned u = __float_as_uint(f);
  return (u & 0x80000000u) ? ~u : (u | 0x80000000u);
}
__device__ __forceinline__ float decF(unsigned u) {
  return __uint_as_float((u & 0x80000000u) ? (u & 0x7fffffffu) : ~u);
}
__device__ __forceinline__ float lrelu(float x) { return x > 0.f ? x : NEG_SLOPE * x; }

// edge e in [0, E+N): real edges then implicit self-loops
__device__ __forceinline__ void edge_sd(const int* __restrict__ ei, int E, int e,
                                        int& s, int& d) {
  if (e < E) { s = ei[e]; d = ei[E + e]; } else { s = d = e - E; }
}

__device__ __forceinline__ v16h pack_a(float4 p0, float4 p1, float4 q0, float4 q1) {
  v16h a;
  a[0]  = (_Float16)p0.x; a[1]  = (_Float16)p0.y; a[2]  = (_Float16)p0.z; a[3]  = (_Float16)p0.w;
  a[4]  = (_Float16)p1.x; a[5]  = (_Float16)p1.y; a[6]  = (_Float16)p1.z; a[7]  = (_Float16)p1.w;
  a[8]  = (_Float16)q0.x; a[9]  = (_Float16)q0.y; a[10] = (_Float16)q0.z; a[11] = (_Float16)q0.w;
  a[12] = (_Float16)q1.x; a[13] = (_Float16)q1.y; a[14] = (_Float16)q1.z; a[15] = (_Float16)q1.w;
  return a;
}

// ---------------------------------------------------------------------------
// GEMM1: xl[N,64] = x[N,128] @ W1[128,64]  via v_wmma_f32_16x16x32_f16
// W1 pre-swizzled into B-fragment order in LDS: [kblk(4)][tile(4)][lane(32)][e(16)]
// so each lane reads its whole fragment with 2x ds_load_b128.
// ---------------------------------------------------------------------------
__global__ __launch_bounds__(256) void gemm1_wmma(const float* __restrict__ x,
                                                  const float* __restrict__ W1,
                                                  float* __restrict__ xl, int N) {
  __shared__ _Float16 Wlds[4 * 4 * 32 * 16];   // 16 KB
  int tid = threadIdx.x;
  for (int i = tid; i < 4 * 4 * 32 * 16; i += 256) {
    int e    = i & 15;
    int lane = (i >> 4) & 31;
    int t    = (i >> 9) & 3;
    int kblk = i >> 11;
    int K    = kblk * 32 + (lane >> 4) * 8 + (e & 7) + (e >> 3) * 16;  // B layout: table order, matches A
    int col  = t * 16 + (lane & 15);
    Wlds[i]  = (_Float16)W1[K * 64 + col];
  }
  __syncthreads();

  int wave = tid >> 5;
  int lane = tid & 31;
  int strip = blockIdx.x * 8 + wave;
  int nStrips = (N + 15) >> 4;
  if (strip >= nStrips) return;
  int rowBase = strip << 4;

  int l    = lane & 15;
  int hi   = lane >> 4;
  int koff = hi * 8;                 // A layout K sub-offset
  int row  = rowBase + l;
  int rowC = row < N ? row : N - 1;  // clamp: unconditional loads, guarded stores
  const float* xrow = x + (size_t)rowC * 128;

  v8f zero = {};
  v8f acc[4];
#pragma unroll
  for (int t = 0; t < 4; ++t) acc[t] = zero;

#pragma unroll
  for (int kblk = 0; kblk < 4; ++kblk) {
    const float4* pa = (const float4*)(xrow + kblk * 32 + koff);
    const float4* pb = (const float4*)(xrow + kblk * 32 + 16 + koff);
    v16h a = pack_a(pa[0], pa[1], pb[0], pb[1]);
#pragma unroll
    for (int t = 0; t < 4; ++t) {
      v16h b = *(const v16h*)&Wlds[(((kblk << 2) + t) * 32 + lane) << 4];
      acc[t] = __builtin_amdgcn_wmma_f32_16x16x32_f16(
          false, a, false, b, (short)0, acc[t], false, false);
    }
  }
  int m0 = hi * 8;
#pragma unroll
  for (int r = 0; r < 8; ++r) {
    int rr = rowBase + m0 + r;
    if (rr < N) {
      size_t base = (size_t)rr * 64;
      xl[base +  0 + l] = acc[0][r];
      xl[base + 16 + l] = acc[1][r];
      xl[base + 32 + l] = acc[2][r];
      xl[base + 48 + l] = acc[3][r];
    }
  }
}

// ---------------------------------------------------------------------------
// GEMM2: hl[N,40] = h[N,64] @ W2[64,40]  (zero-padded to 48 cols, 3 tiles)
// ---------------------------------------------------------------------------
__global__ __launch_bounds__(256) void gemm2_wmma(const float* __restrict__ h,
                                                  const float* __restrict__ W2,
                                                  float* __restrict__ hl, int N) {
  __shared__ _Float16 Wlds[2 * 3 * 32 * 16];   // 6 KB
  int tid = threadIdx.x;
  for (int i = tid; i < 2 * 3 * 32 * 16; i += 256) {
    int e    = i & 15;
    int r    = i >> 4;
    int lane = r & 31;
    int q    = r >> 5;
    int t    = q % 3;
    int kblk = q / 3;
    int K    = kblk * 32 + (lane >> 4) * 8 + (e & 7) + (e >> 3) * 16;
    int col  = t * 16 + (lane & 15);
    Wlds[i]  = (col < 40) ? (_Float16)W2[K * 40 + col] : (_Float16)0.f;
  }
  __syncthreads();

  int wave = tid >> 5;
  int lane = tid & 31;
  int strip = blockIdx.x * 8 + wave;
  int nStrips = (N + 15) >> 4;
  if (strip >= nStrips) return;
  int rowBase = strip << 4;

  int l    = lane & 15;
  int hi   = lane >> 4;
  int koff = hi * 8;
  int row  = rowBase + l;
  int rowC = row < N ? row : N - 1;
  const float* hrow = h + (size_t)rowC * 64;

  v8f zero = {};
  v8f acc[3];
#pragma unroll
  for (int t = 0; t < 3; ++t) acc[t] = zero;

#pragma unroll
  for (int kblk = 0; kblk < 2; ++kblk) {
    const float4* pa = (const float4*)(hrow + kblk * 32 + koff);
    const float4* pb = (const float4*)(hrow + kblk * 32 + 16 + koff);
    v16h a = pack_a(pa[0], pa[1], pb[0], pb[1]);
#pragma unroll
    for (int t = 0; t < 3; ++t) {
      v16h b = *(const v16h*)&Wlds[((kblk * 3 + t) * 32 + lane) << 4];
      acc[t] = __builtin_amdgcn_wmma_f32_16x16x32_f16(
          false, a, false, b, (short)0, acc[t], false, false);
    }
  }
  int m0 = hi * 8;
#pragma unroll
  for (int t = 0; t < 3; ++t) {
    int col = t * 16 + l;
    if (col < 40) {
#pragma unroll
      for (int r = 0; r < 8; ++r) {
        int rr = rowBase + m0 + r;
        if (rr < N) hl[(size_t)rr * 40 + col] = acc[t][r];
      }
    }
  }
}

// ---------------------------------------------------------------------------
// Layer 1 node/edge kernels (H=8, C=8)
// ---------------------------------------------------------------------------
__global__ void alpha1_init(const float* __restrict__ xl,
                            const float* __restrict__ a_src,
                            const float* __restrict__ a_dst,
                            float* __restrict__ as, float* __restrict__ ad,
                            unsigned* __restrict__ amax, float* __restrict__ denom,
                            float* __restrict__ accum, int N) {
  int gid = blockIdx.x * blockDim.x + threadIdx.x;
  if (gid >= N * 8) return;
  int n = gid >> 3, hh = gid & 7;
  const float4* v = (const float4*)(xl + (size_t)n * 64 + hh * 8);
  float4 v0 = v[0], v1 = v[1];
  const float4* ws = (const float4*)(a_src + hh * 8);
  const float4* wd = (const float4*)(a_dst + hh * 8);
  float4 s0 = ws[0], s1 = ws[1], d0 = wd[0], d1 = wd[1];
  float ssum = v0.x * s0.x + v0.y * s0.y + v0.z * s0.z + v0.w * s0.w +
               v1.x * s1.x + v1.y * s1.y + v1.z * s1.z + v1.w * s1.w;
  float dsum = v0.x * d0.x + v0.y * d0.y + v0.z * d0.z + v0.w * d0.w +
               v1.x * d1.x + v1.y * d1.y + v1.z * d1.z + v1.w * d1.w;
  as[gid] = ssum; ad[gid] = dsum;
  amax[gid] = ENC_NEG_INF;
  denom[gid] = 0.f;
  float4* acc = (float4*)(accum + (size_t)n * 64 + hh * 8);
  float4 z = {0.f, 0.f, 0.f, 0.f};
  acc[0] = z; acc[1] = z;
}

__global__ void edge_max1(const int* __restrict__ ei, const float* __restrict__ as,
                          const float* __restrict__ ad, unsigned* __restrict__ amax,
                          int E, int ET) {
  int e = blockIdx.x * blockDim.x + threadIdx.x;
  if (e >= ET) return;
  int s, d; edge_sd(ei, E, e, s, d);
  const float4* ps = (const float4*)(as + (size_t)s * 8);
  const float4* pd = (const float4*)(ad + (size_t)d * 8);
  float4 s0 = ps[0], s1 = ps[1], d0 = pd[0], d1 = pd[1];
  float sv[8] = {s0.x, s0.y, s0.z, s0.w, s1.x, s1.y, s1.z, s1.w};
  float dv[8] = {d0.x, d0.y, d0.z, d0.w, d1.x, d1.y, d1.z, d1.w};
  unsigned* am = amax + (size_t)d * 8;
#pragma unroll
  for (int h = 0; h < 8; ++h) atomicMax(&am[h], encF(lrelu(sv[h] + dv[h])));
}

__global__ void edge_sum1(const int* __restrict__ ei, const float* __restrict__ as,
                          const float* __restrict__ ad, const unsigned* __restrict__ amax,
                          float* __restrict__ denom, int E, int ET) {
  int e = blockIdx.x * blockDim.x + threadIdx.x;
  if (e >= ET) return;
  int s, d; edge_sd(ei, E, e, s, d);
  const float4* ps = (const float4*)(as + (size_t)s * 8);
  const float4* pd = (const float4*)(ad + (size_t)d * 8);
  float4 s0 = ps[0], s1 = ps[1], d0 = pd[0], d1 = pd[1];
  float sv[8] = {s0.x, s0.y, s0.z, s0.w, s1.x, s1.y, s1.z, s1.w};
  float dv[8] = {d0.x, d0.y, d0.z, d0.w, d1.x, d1.y, d1.z, d1.w};
  const unsigned* am = amax + (size_t)d * 8;
  float* dn = denom + (size_t)d * 8;
#pragma unroll
  for (int h = 0; h < 8; ++h)
    atomicAdd(&dn[h], expf(lrelu(sv[h] + dv[h]) - decF(am[h])));
}

// one thread per (edge, head): attn computed once, 8-wide vector load, 8 atomics
__global__ void scatter1(const int* __restrict__ ei, const float* __restrict__ xl,
                         const float* __restrict__ as, const float* __restrict__ ad,
                         const unsigned* __restrict__ amax,
                         const float* __restrict__ denom,
                         float* __restrict__ accum, int E, int ET) {
  int gid = blockIdx.x * blockDim.x + threadIdx.x;
  if (gid >= ET * 8) return;
  int e = gid >> 3, h = gid & 7;
  int s, d; edge_sd(ei, E, e, s, d);
  float al = lrelu(as[(size_t)s * 8 + h] + ad[(size_t)d * 8 + h]);
  float ea = expf(al - decF(amax[(size_t)d * 8 + h]));
  float attn = ea / (denom[(size_t)d * 8 + h] + EPS_DEN);
  const float4* xs = (const float4*)(xl + (size_t)s * 64 + h * 8);
  float4 m0 = xs[0], m1 = xs[1];
  float* ap = accum + (size_t)d * 64 + h * 8;
  atomicAdd(&ap[0], m0.x * attn); atomicAdd(&ap[1], m0.y * attn);
  atomicAdd(&ap[2], m0.z * attn); atomicAdd(&ap[3], m0.w * attn);
  atomicAdd(&ap[4], m1.x * attn); atomicAdd(&ap[5], m1.y * attn);
  atomicAdd(&ap[6], m1.z * attn); atomicAdd(&ap[7], m1.w * attn);
}

__global__ void finish1(float* __restrict__ accum, const float* __restrict__ b1, int N) {
  int gid = blockIdx.x * blockDim.x + threadIdx.x;
  if (gid >= N * 64) return;
  float v = accum[gid] + b1[gid & 63];
  accum[gid] = v > 0.f ? v : (expf(v) - 1.f);   // ELU, alpha=1
}

// ---------------------------------------------------------------------------
// Layer 2 node/edge kernels (H=1, C=40) — scatter straight into d_out
// ---------------------------------------------------------------------------
__global__ void alpha2_init(const float* __restrict__ hl,
                            const float* __restrict__ a_src,
                            const float* __restrict__ a_dst,
                            float* __restrict__ as, float* __restrict__ ad,
                            unsigned* __restrict__ amax, float* __restrict__ denom,
                            float* __restrict__ out, int N) {
  int n = blockIdx.x * blockDim.x + threadIdx.x;
  if (n >= N) return;
  const float4* v = (const float4*)(hl + (size_t)n * 40);
  float s1 = 0.f, s2 = 0.f;
#pragma unroll
  for (int q = 0; q < 10; ++q) {
    float4 vv = v[q];
    const float4 sa = ((const float4*)a_src)[q];
    const float4 da = ((const float4*)a_dst)[q];
    s1 += vv.x * sa.x + vv.y * sa.y + vv.z * sa.z + vv.w * sa.w;
    s2 += vv.x * da.x + vv.y * da.y + vv.z * da.z + vv.w * da.w;
  }
  as[n] = s1; ad[n] = s2;
  amax[n] = ENC_NEG_INF;
  denom[n] = 0.f;
  float4* o = (float4*)(out + (size_t)n * 40);
  float4 z = {0.f, 0.f, 0.f, 0.f};
#pragma unroll
  for (int q = 0; q < 10; ++q) o[q] = z;
}

__global__ void edge_max2(const int* __restrict__ ei, const float* __restrict__ as,
                          const float* __restrict__ ad, unsigned* __restrict__ amax,
                          int E, int ET) {
  int e = blockIdx.x * blockDim.x + threadIdx.x;
  if (e >= ET) return;
  int s, d; edge_sd(ei, E, e, s, d);
  atomicMax(&amax[d], encF(lrelu(as[s] + ad[d])));
}

__global__ void edge_sum2(const int* __restrict__ ei, const float* __restrict__ as,
                          const float* __restrict__ ad, const unsigned* __restrict__ amax,
                          float* __restrict__ denom, int E, int ET) {
  int e = blockIdx.x * blockDim.x + threadIdx.x;
  if (e >= ET) return;
  int s, d; edge_sd(ei, E, e, s, d);
  atomicAdd(&denom[d], expf(lrelu(as[s] + ad[d]) - decF(amax[d])));
}

// one thread per edge: attn once, 10x float4 loads, 40 atomics
__global__ void scatter2(const int* __restrict__ ei, const float* __restrict__ hl,
                         const float* __restrict__ as, const float* __restrict__ ad,
                         const unsigned* __restrict__ amax,
                         const float* __restrict__ denom,
                         float* __restrict__ out, int E, int ET) {
  int e = blockIdx.x * blockDim.x + threadIdx.x;
  if (e >= ET) return;
  int s, d; edge_sd(ei, E, e, s, d);
  float ea = expf(lrelu(as[s] + ad[d]) - decF(amax[d]));
  float attn = ea / (denom[d] + EPS_DEN);
  const float4* hs = (const float4*)(hl + (size_t)s * 40);
  float* op = out + (size_t)d * 40;
#pragma unroll
  for (int q = 0; q < 10; ++q) {
    float4 m = hs[q];
    atomicAdd(&op[q * 4 + 0], m.x * attn);
    atomicAdd(&op[q * 4 + 1], m.y * attn);
    atomicAdd(&op[q * 4 + 2], m.z * attn);
    atomicAdd(&op[q * 4 + 3], m.w * attn);
  }
}

__global__ void final2(float* __restrict__ out, const float* __restrict__ b2, int N) {
  int gid = blockIdx.x * blockDim.x + threadIdx.x;
  if (gid >= N * 40) return;
  out[gid] += b2[gid % 40];
}

// ---------------------------------------------------------------------------
extern "C" void kernel_launch(void* const* d_in, const int* in_sizes, int n_in,
                              void* d_out, int out_size, void* d_ws, size_t ws_size,
                              hipStream_t stream) {
  const float* x      = (const float*)d_in[0];
  const int*   ei     = (const int*)d_in[1];   // [2,E] int32
  const float* W1     = (const float*)d_in[2];
  const float* a_src1 = (const float*)d_in[3];
  const float* a_dst1 = (const float*)d_in[4];
  const float* b1     = (const float*)d_in[5];
  const float* W2     = (const float*)d_in[6];
  const float* a_src2 = (const float*)d_in[7];
  const float* a_dst2 = (const float*)d_in[8];
  const float* b2     = (const float*)d_in[9];
  float* out = (float*)d_out;

  const int N  = in_sizes[0] / 128;
  const int E  = in_sizes[1] / 2;
  const int ET = E + N;          // with self-loops

  // workspace layout (floats): xl/hl | accum(h) | as | ad | amax | denom
  float*    xl    = (float*)d_ws;                       // N*64 (reused as hl: N*40)
  float*    accum = xl    + (size_t)N * 64;             // N*64 (becomes h)
  float*    as1   = accum + (size_t)N * 64;             // N*8  (reused layer 2)
  float*    ad1   = as1   + (size_t)N * 8;              // N*8
  unsigned* amax  = (unsigned*)(ad1 + (size_t)N * 8);   // N*8
  float*    denom = (float*)(amax + (size_t)N * 8);     // N*8
  float*    hl    = xl;

  auto nb = [](long long n, int b) { return (unsigned)((n + b - 1) / b); };
  const int nStrips = (N + 15) >> 4;

  // -------- layer 1 --------
  gemm1_wmma<<<nb(nStrips, 8), 256, 0, stream>>>(x, W1, xl, N);
  alpha1_init<<<nb((long long)N * 8, 256), 256, 0, stream>>>(
      xl, a_src1, a_dst1, as1, ad1, amax, denom, accum, N);
  edge_max1<<<nb(ET, 256), 256, 0, stream>>>(ei, as1, ad1, amax, E, ET);
  edge_sum1<<<nb(ET, 256), 256, 0, stream>>>(ei, as1, ad1, amax, denom, E, ET);
  scatter1<<<nb((long long)ET * 8, 256), 256, 0, stream>>>(
      ei, xl, as1, ad1, amax, denom, accum, E, ET);
  finish1<<<nb((long long)N * 64, 256), 256, 0, stream>>>(accum, b1, N);

  // -------- layer 2 --------
  gemm2_wmma<<<nb(nStrips, 8), 256, 0, stream>>>(accum, W2, hl, N);
  alpha2_init<<<nb(N, 256), 256, 0, stream>>>(
      hl, a_src2, a_dst2, as1, ad1, amax, denom, out, N);
  edge_max2<<<nb(ET, 256), 256, 0, stream>>>(ei, as1, ad1, amax, E, ET);
  edge_sum2<<<nb(ET, 256), 256, 0, stream>>>(ei, as1, ad1, amax, denom, E, ET);
  scatter2<<<nb(ET, 256), 256, 0, stream>>>(ei, hl, as1, ad1, amax, denom, out, E, ET);
  final2<<<nb((long long)N * 40, 256), 256, 0, stream>>>(out, b2, N);
}
